// Nonlocal_Spatial_Multihead_Self_Attention_14190571946268
// MI455X (gfx1250) — matrix operand order repaired and checked by reading, not compiled, not run on hardware
//
#include <hip/hip_runtime.h>

// ---------------------------------------------------------------------------
// Nonlocal spatial windowed multi-head attention for MI455X (gfx1250, wave32).
// Memory-bound workload (~55 GFLOP vs ~0.7 GB min traffic): f16 intermediates,
// v_wmma_f32_16x16x32_f16 everywhere, LDS-staged fragments, pipelined b-frags
// (each feeds 2 WMMAs), LDS-transposed f32 epilogue for coalesced NCHW stores,
// and async global->LDS copies where the data is a pure f16 copy.
// ---------------------------------------------------------------------------

typedef _Float16 f16;
typedef _Float16 v16h __attribute__((ext_vector_type(16)));
typedef _Float16 v8h  __attribute__((ext_vector_type(8)));
typedef _Float16 v2h  __attribute__((ext_vector_type(2)));
typedef float    v8f  __attribute__((ext_vector_type(8)));

#if __has_builtin(__builtin_amdgcn_global_load_async_to_lds_b128) && \
    __has_builtin(__builtin_amdgcn_s_wait_asynccnt)
#define HAVE_ASYNC_LDS 1
// Toolchain signature (from clang diagnostic): pointee is a 16B int vector,
// src in address_space(1) (global), dst in address_space(3) (LDS).
typedef int v4i __attribute__((vector_size(16)));
typedef __attribute__((address_space(1))) v4i g_v4i;
typedef __attribute__((address_space(3))) v4i l_v4i;
#endif

#define BATCH   128
#define CDIM    224      // dim == inner
#define HDIM    32
#define WDIM    32
#define HW      1024     // 32*32
#define HEADS   8
#define DH      28       // head dim
#define NWIN    16       // windows (attention length)
#define MPOS    64       // positions per window
#define MTOT    (BATCH*HW)   // 131072 rows
#define MT      128      // GEMM M tile
#define NTILES  14       // 224/16
#define KCH     7        // 224/32

static __device__ __forceinline__ v16h cat16(v8h lo, v8h hi) {
  return __builtin_shufflevector(lo, hi, 0,1,2,3,4,5,6,7,8,9,10,11,12,13,14,15);
}
static __device__ __forceinline__ v8f zf8() {
  v8f z;
#pragma unroll
  for (int i = 0; i < 8; ++i) z[i] = 0.0f;
  return z;
}
static __device__ __forceinline__ v8h zh8() {
  v8h z;
#pragma unroll
  for (int i = 0; i < 8; ++i) z[i] = (f16)0.0f;
  return z;
}
// B fragment (32x16, K = (j>>3)*16 + (lane>>4)*8 + (j&7)) from transposed [N][K] LDS image.
static __device__ __forceinline__ v16h loadB(const f16 (*B)[CDIM], int col,
                                             int kb, int hi) {
  return cat16(*(const v8h*)&B[col][kb + hi * 8],
               *(const v8h*)&B[col][kb + 16 + hi * 8]);
}
// A fragment (16x32, K = (j>>3)*16 + (lane>>4)*8 + (j&7)) from [row][K] LDS.
static __device__ __forceinline__ v16h loadA(const f16 (*A)[CDIM], int row,
                                             int kb, int hi) {
  return cat16(*(const v8h*)&A[row][kb + hi * 8],
               *(const v8h*)&A[row][kb + 16 + hi * 8]);
}

// ---------------------------------------------------------------------------
// Kernel 1: Q/K/V projections.  C[M,224] = x^T[M,224] @ W[224,224] for 3 W's.
// 8 waves: wave = (strip-pair sp, n-half). acc[2][7]; each b-frag -> 2 WMMAs.
// ---------------------------------------------------------------------------
__global__ __launch_bounds__(256) void qkv_gemm_kernel(
    const float* __restrict__ x,
    const float* __restrict__ Wq, const float* __restrict__ Wk,
    const float* __restrict__ Wv,
    f16* __restrict__ Qs, f16* __restrict__ Ks, f16* __restrict__ Vs)
{
  __shared__ f16 Alds[MT][CDIM];        // 57344 B, [row][k]
  __shared__ f16 Blds[CDIM][CDIM];      // 100352 B, transposed [n][k]

  const int tid  = threadIdx.x;
  const int lane = tid & 31;
  const int wid  = tid >> 5;
  const int hi   = lane >> 4;
  const int sp   = wid >> 1;                   // strip pair: rows sp*32+{0,16}
  const int nh0  = (wid & 1) * 7;              // 7 of the 14 n-tiles
  const int mblk = blockIdx.x;                 // 1024 blocks
  const long row0 = (long)mblk * MT;
  const int  b    = mblk >> 3;                 // 8 tiles per batch image
  const int  hw0  = (mblk & 7) * MT;

  // Stage A: A[r][k] = x[b][k][hw0+r]  (coalesced along r, f32 -> f16)
  {
    const float* xb = x + (size_t)b * CDIM * HW;
    const int r  = tid & 127;
    const int kh = tid >> 7;                   // 0 or 1
#pragma unroll 4
    for (int k2 = 0; k2 < CDIM / 2; ++k2) {
      const int k = k2 * 2 + kh;
      Alds[r][k] = (f16)xb[(size_t)k * HW + hw0 + r];
    }
  }

  const float* Wmats[3] = { Wq, Wk, Wv };
  f16*         Outs[3]  = { Qs, Ks, Vs };

  for (int w = 0; w < 3; ++w) {
    __syncthreads();   // A ready (iter 0); Blds free of previous readers
    {
      const float* Wm = Wmats[w];
      if (tid < CDIM) {
        for (int k = 0; k < CDIM; ++k)
          Blds[tid][k] = (f16)Wm[(size_t)k * CDIM + tid];
      }
    }
    __syncthreads();

    v8f acc[2][7];
#pragma unroll
    for (int st = 0; st < 2; ++st)
#pragma unroll
      for (int t = 0; t < 7; ++t) acc[st][t] = zf8();

    const int arow = sp * 32 + (lane & 15);
#pragma unroll
    for (int kc = 0; kc < KCH; ++kc) {
      const int kb = kc * 32;
      const v16h af0 = loadA(Alds, arow,      kb, hi);
      const v16h af1 = loadA(Alds, arow + 16, kb, hi);
      v16h bf = loadB(Blds, (nh0 + 0) * 16 + (lane & 15), kb, hi);
#pragma unroll
      for (int t = 0; t < 7; ++t) {
        v16h bn;
        if (t < 6) bn = loadB(Blds, (nh0 + t + 1) * 16 + (lane & 15), kb, hi);
        acc[0][t] = __builtin_amdgcn_wmma_f32_16x16x32_f16(
            false, af0, false, bf, (short)0, acc[0][t], false, false);
        acc[1][t] = __builtin_amdgcn_wmma_f32_16x16x32_f16(
            false, af1, false, bf, (short)0, acc[1][t], false, false);
        if (t < 6) bf = bn;
      }
    }

    // Store row-major f16 (16 lanes -> 32B contiguous per row segment)
    f16* Om = Outs[w];
#pragma unroll
    for (int st = 0; st < 2; ++st)
#pragma unroll
      for (int t = 0; t < 7; ++t) {
        const int nc = (nh0 + t) * 16 + (lane & 15);
#pragma unroll
        for (int r = 0; r < 8; ++r) {
          const long grow = row0 + sp * 32 + st * 16 + r + hi * 8;
          Om[(size_t)grow * CDIM + nc] = (f16)acc[st][t][r];
        }
      }
  }
}

// ---------------------------------------------------------------------------
// Kernel 2: windowed attention.  One block per (b, m); wave w = head h.
// sim[16,16] = scale*(q[16,28] k[16,28]^T) + pos ; softmax ; o = attn v[16,28]
// ---------------------------------------------------------------------------
__global__ __launch_bounds__(256) void attn_kernel(
    const f16* __restrict__ Qs, const f16* __restrict__ Ks,
    const f16* __restrict__ Vs, const float* __restrict__ pos,
    const float* __restrict__ scale, f16* __restrict__ Os)
{
  __shared__ f16 qa[HEADS][16][32];     // [n][d pad 32]
  __shared__ f16 ka[HEADS][16][32];     // [n][d pad 32]
  __shared__ f16 vt[HEADS][32][16];     // [d pad 32][n]
  __shared__ f16 sattn[HEADS][16][16];  // [i][j]

  const int tid  = threadIdx.x;
  const int lane = tid & 31;
  const int h    = tid >> 5;            // head == wave id
  const int hi   = lane >> 4;
  const int col  = lane & 15;

  const int bm = blockIdx.x;            // b*64 + m
  const int b  = bm >> 6;
  const int m  = bm & 63;
  const int wsr = m >> 3, wsc = m & 7;
  const float sc = scale[0];

  // ---- stage q, k, v for this (b, m, h): 16 windows x 28 dims (u32 pairs)
  for (int idx = lane; idx < NWIN * (DH / 2); idx += 32) {   // 224 chunks
    const int n = idx / 14;
    const int j = idx - n * 14;
    const int nh = n >> 2, nw = n & 3;
    const int hwi = (nh * 8 + wsr) * WDIM + (nw * 8 + wsc);
    const size_t base = ((size_t)(b * HW + hwi)) * CDIM + h * DH + j * 2;
    *(v2h*)&qa[h][n][j * 2] = *(const v2h*)(Qs + base);
    *(v2h*)&ka[h][n][j * 2] = *(const v2h*)(Ks + base);
    const v2h vv = *(const v2h*)(Vs + base);
    vt[h][j * 2 + 0][n] = vv[0];
    vt[h][j * 2 + 1][n] = vv[1];
  }
  // zero the K-dim padding (d = 28..31)
  {
    v2h z2; z2[0] = (f16)0.0f; z2[1] = (f16)0.0f;
    if (lane < 16) {
      *(v2h*)&qa[h][lane][28] = z2;  *(v2h*)&qa[h][lane][30] = z2;
      *(v2h*)&ka[h][lane][28] = z2;  *(v2h*)&ka[h][lane][30] = z2;
    }
    *(v2h*)(&vt[h][28][0] + lane * 2) = z2;   // rows 28..31 of vt (64 halfs)
  }

  // ---- sim = q @ k^T  (one WMMA, K = 32 with zero pad)
  const v16h afq = cat16(*(const v8h*)&qa[h][col][hi * 8],
                         *(const v8h*)&qa[h][col][16 + hi * 8]);
  const v16h bfk = cat16(*(const v8h*)&ka[h][col][hi * 8],
                         *(const v8h*)&ka[h][col][16 + hi * 8]);
  v8f sim = zf8();
  sim = __builtin_amdgcn_wmma_f32_16x16x32_f16(
      false, afq, false, bfk, (short)0, sim, false, false);

  // ---- scale + pos bias + row softmax (rows live across 16-lane halves)
#pragma unroll
  for (int r = 0; r < 8; ++r) {
    const int mi = r + hi * 8;
    float v = sim[r] * sc + pos[((size_t)h * 16 + mi) * 16 + col];
    float mx = v;
#pragma unroll
    for (int d = 1; d < 16; d <<= 1) mx = fmaxf(mx, __shfl_xor(mx, d, 32));
    const float e = __expf(v - mx);
    float s = e;
#pragma unroll
    for (int d = 1; d < 16; d <<= 1) s += __shfl_xor(s, d, 32);
    sattn[h][mi][col] = (f16)(e / s);
  }

  // ---- o = attn @ v  (A K=16..31 zeroed so B pad lanes are don't-care)
  const v16h afa = cat16(*(const v8h*)&sattn[h][col][hi * 8], zh8());
#pragma unroll
  for (int dt = 0; dt < 2; ++dt) {
    const int dcol = dt * 16 + col;
    const v16h bfv = cat16(*(const v8h*)&vt[h][dcol][hi * 8],
                           *(const v8h*)&vt[h][dcol][8]);
    v8f oacc = zf8();
    oacc = __builtin_amdgcn_wmma_f32_16x16x32_f16(
        false, afa, false, bfv, (short)0, oacc, false, false);
#pragma unroll
    for (int r = 0; r < 8; ++r) {
      const int ni = r + hi * 8;
      const int d  = dt * 16 + col;
      if (d < DH) {
        Os[(((size_t)(b * NWIN + ni)) * MPOS + m) * CDIM + h * DH + d] =
            (f16)oacc[r];
      }
    }
  }
}

// ---------------------------------------------------------------------------
// Kernel 3: output projection.  out = O[M,224] @ Wo + bo.
// A tile staged with async global->LDS copies (pure f16 copy).  Epilogue
// transposes through LDS (stride 225 to avoid bank conflicts) and emits
// 32B float4 stores along W for the NCHW output.
// ---------------------------------------------------------------------------
__global__ __launch_bounds__(256) void out_gemm_kernel(
    const f16* __restrict__ Os, const float* __restrict__ Wo,
    const float* __restrict__ bo, float* __restrict__ out)
{
  __shared__ __align__(16) unsigned char smem[MT * CDIM * 2 + CDIM * CDIM * 2];
  f16 (*Alds)[CDIM] = reinterpret_cast<f16(*)[CDIM]>(smem);
  f16 (*Blds)[CDIM] = reinterpret_cast<f16(*)[CDIM]>(smem + MT * CDIM * 2);
  float (*Clds)[CDIM + 1] =
      reinterpret_cast<float(*)[CDIM + 1]>(smem);   // epilogue overlay, 57600B

  const int tid  = threadIdx.x;
  const int lane = tid & 31;
  const int wid  = tid >> 5;
  const int hi   = lane >> 4;
  const int sp   = wid >> 1;
  const int nh0  = (wid & 1) * 7;
  const int mblk = blockIdx.x;
  const long row0 = (long)mblk * MT;

  // ---- Stage A (pure f16 copy: use CDNA5 async global->LDS when available)
#if defined(HAVE_ASYNC_LDS)
  {
    const f16* gsrc = Os + (size_t)row0 * CDIM;
    f16* lbase = &Alds[0][0];
#pragma unroll 2
    for (int idx = tid; idx < MT * CDIM / 8; idx += 256) {   // 16B chunks
      v4i* gp_gen = (v4i*)(gsrc + (size_t)idx * 8);          // drop const
      v4i* lp_gen = (v4i*)(lbase + (size_t)idx * 8);
      __builtin_amdgcn_global_load_async_to_lds_b128(
          (g_v4i*)gp_gen, (l_v4i*)lp_gen, 0, 0);
    }
    __builtin_amdgcn_s_wait_asynccnt(0);
  }
#else
  for (int idx = tid; idx < MT * (CDIM / 2); idx += 256) {
    const int r = idx / 112;
    const int j = idx - r * 112;
    *(v2h*)&Alds[r][j * 2] =
        *(const v2h*)(Os + (size_t)(row0 + r) * CDIM + j * 2);
  }
#endif
  // ---- Stage B transposed
  if (tid < CDIM) {
    for (int k = 0; k < CDIM; ++k)
      Blds[tid][k] = (f16)Wo[(size_t)k * CDIM + tid];
  }
  __syncthreads();

  v8f acc[2][7];
#pragma unroll
  for (int st = 0; st < 2; ++st)
#pragma unroll
    for (int t = 0; t < 7; ++t) acc[st][t] = zf8();

  const int arow = sp * 32 + (lane & 15);
#pragma unroll
  for (int kc = 0; kc < KCH; ++kc) {
    const int kb = kc * 32;
    const v16h af0 = loadA(Alds, arow,      kb, hi);
    const v16h af1 = loadA(Alds, arow + 16, kb, hi);
    v16h bf = loadB(Blds, (nh0 + 0) * 16 + (lane & 15), kb, hi);
#pragma unroll
    for (int t = 0; t < 7; ++t) {
      v16h bn;
      if (t < 6) bn = loadB(Blds, (nh0 + t + 1) * 16 + (lane & 15), kb, hi);
      acc[0][t] = __builtin_amdgcn_wmma_f32_16x16x32_f16(
          false, af0, false, bf, (short)0, acc[0][t], false, false);
      acc[1][t] = __builtin_amdgcn_wmma_f32_16x16x32_f16(
          false, af1, false, bf, (short)0, acc[1][t], false, false);
      if (t < 6) bf = bn;
    }
  }

  __syncthreads();   // compute done; A/B LDS may now be overlaid by Clds

  // ---- Epilogue: two 64-row halves; bias + transpose through LDS, then
  //      32B-coalesced float4 stores into NCHW output.
#pragma unroll
  for (int hf = 0; hf < 2; ++hf) {
    if ((sp >> 1) == hf) {
#pragma unroll
      for (int st = 0; st < 2; ++st)
#pragma unroll
        for (int t = 0; t < 7; ++t) {
          const int c = (nh0 + t) * 16 + (lane & 15);
          const float bias = bo[c];
#pragma unroll
          for (int r = 0; r < 8; ++r) {
            const int lr = (sp & 1) * 32 + st * 16 + r + hi * 8;
            Clds[lr][c] = acc[st][t][r] + bias;
          }
        }
    }
    __syncthreads();

    const long grow0 = row0 + hf * 64;       // 64 rows == one window n
    const int bb = (int)(grow0 >> 10);
    const int n  = (int)((grow0 >> 6) & 15);
    const int nh = n >> 2, nw = n & 3;
    for (int s = tid; s < CDIM * 8; s += 256) {
      const int c   = s >> 3;
      const int wsr = s & 7;
      float tmp[8];
#pragma unroll
      for (int wsc = 0; wsc < 8; ++wsc) tmp[wsc] = Clds[wsr * 8 + wsc][c];
      float* dst =
          &out[(((size_t)bb * CDIM + c) * HDIM + nh * 8 + wsr) * WDIM + nw * 8];
      *reinterpret_cast<float4*>(dst)     = make_float4(tmp[0], tmp[1], tmp[2], tmp[3]);
      *(reinterpret_cast<float4*>(dst) + 1) = make_float4(tmp[4], tmp[5], tmp[6], tmp[7]);
    }
    __syncthreads();
  }
}

// ---------------------------------------------------------------------------
extern "C" void kernel_launch(void* const* d_in, const int* in_sizes, int n_in,
                              void* d_out, int out_size, void* d_ws,
                              size_t ws_size, hipStream_t stream) {
  const float* x     = (const float*)d_in[0];
  const float* Wq    = (const float*)d_in[1];
  const float* Wk    = (const float*)d_in[2];
  const float* Wv    = (const float*)d_in[3];
  const float* Wo    = (const float*)d_in[4];
  const float* bo    = (const float*)d_in[5];
  const float* pos   = (const float*)d_in[6];
  const float* scale = (const float*)d_in[7];

  const size_t SZ = (size_t)MTOT * CDIM;   // elements per f16 scratch matrix
  f16* Qs = (f16*)d_ws;
  f16* Ks = Qs + SZ;
  f16* Vs = Ks + SZ;
  f16* Os = Vs + SZ;

  qkv_gemm_kernel<<<MTOT / MT, 256, 0, stream>>>(x, Wq, Wk, Wv, Qs, Ks, Vs);
  attn_kernel<<<BATCH * MPOS, 256, 0, stream>>>(Qs, Ks, Vs, pos, scale, Os);
  out_gemm_kernel<<<MTOT / MT, 256, 0, stream>>>(Os, Wo, bo, (float*)d_out);
}
